// AttentionLayer_10582799417402
// MI455X (gfx1250) — compile-verified, matrix operand order checked by reading, not run on hardware
//
#include <hip/hip_runtime.h>
#include <hip/hip_bf16.h>
#include <stdint.h>

// ---------------------------------------------------------------------------
// Problem constants (from reference)
// ---------------------------------------------------------------------------
#define Bc    4
#define Lc    4096
#define Dc    1024
#define Hc    16
#define DKc   64
#define Rc    20
#define LENc  200
#define Mrows (Bc * Lc)     // 16384
#define Kdim  Dc            // 1024
#define Ndim  (Hc * DKc)    // 1024

// ---------------------------------------------------------------------------
// Types & helpers
// ---------------------------------------------------------------------------
typedef __attribute__((ext_vector_type(16))) __bf16 v16bf;
typedef __attribute__((ext_vector_type(8)))  float  v8f;
typedef __attribute__((ext_vector_type(4)))  int    v4i_t;

union FragAB { v16bf v; uint4 q[2]; };

#define AS1 __attribute__((address_space(1)))
#define AS3 __attribute__((address_space(3)))

// CDNA5 async global->LDS staging (ASYNCcnt-tracked), guarded so the file
// still compiles if the toolchain lacks the builtins.
#if defined(__has_builtin)
#  if __has_builtin(__builtin_amdgcn_global_load_async_to_lds_b128)
#    define HAVE_ASYNC_LDS 1
#  endif
#endif
#ifndef HAVE_ASYNC_LDS
#  define HAVE_ASYNC_LDS 0
#endif

__device__ __forceinline__ void stage16(const unsigned short* g,
                                        unsigned short* l) {
#if HAVE_ASYNC_LDS
  // per-lane 16B global -> LDS, bypasses VGPRs, increments ASYNCcnt
  __builtin_amdgcn_global_load_async_to_lds_b128(
      (AS1 v4i_t*)(uintptr_t)g, (AS3 v4i_t*)(uintptr_t)l, 0, 0);
#else
  *(uint4*)l = *(const uint4*)g;
#endif
}

__device__ __forceinline__ void stage_wait() {
#if HAVE_ASYNC_LDS
#  if __has_builtin(__builtin_amdgcn_s_wait_asynccnt)
  __builtin_amdgcn_s_wait_asynccnt(0);
#  else
  asm volatile("s_wait_asynccnt 0x0" ::: "memory");
#  endif
#endif
}

__device__ __forceinline__ unsigned short f2bf(float f) {
  unsigned int u = __float_as_uint(f);
  u += 0x7FFFu + ((u >> 16) & 1u);          // round-to-nearest-even
  return (unsigned short)(u >> 16);
}
__device__ __forceinline__ float bf2f(unsigned short h) {
  return __uint_as_float(((unsigned int)h) << 16);
}

// ---------------------------------------------------------------------------
// Conversion kernels
// ---------------------------------------------------------------------------
__global__ void cvt_f32_bf16(const float* __restrict__ in,
                             unsigned short* __restrict__ out, int n) {
  int i = blockIdx.x * 256 + threadIdx.x;
  if (i < n) out[i] = f2bf(in[i]);
}

// W: K x N row-major f32  ->  WT: N x K row-major bf16
__global__ void cvt_transpose(const float* __restrict__ W,
                              unsigned short* __restrict__ WT, int K, int N) {
  int i = blockIdx.x * 256 + threadIdx.x;
  if (i < K * N) {
    int nn = i / K, kk = i % K;
    WT[i] = f2bf(W[(size_t)kk * N + nn]);
  }
}

// ---------------------------------------------------------------------------
// Landmark low-rank projections:  we/wr (B,H,DK,R)
//   xl[b,h,d,l] = x[b, idx[l], h*64+d],  idx[l] = floor(l*(L-1)/(LEN-1))
//   we[b,h,d,e] = sum_l xl * We[h,l,e]
// ---------------------------------------------------------------------------
__global__ void landmark_kernel(const float* __restrict__ x,
                                const float* __restrict__ We,
                                const float* __restrict__ Wr,
                                float* __restrict__ we,
                                float* __restrict__ wr) {
  int t = blockIdx.x * 256 + threadIdx.x;
  if (t >= Bc * Hc * DKc * Rc) return;
  int e  = t % Rc;
  int d  = (t / Rc) % DKc;
  int h  = (t / (Rc * DKc)) % Hc;
  int bb = t / (Rc * DKc * Hc);
  float aE = 0.f, aR = 0.f;
  for (int l = 0; l < LENc; ++l) {
    int il = (l * (Lc - 1)) / (LENc - 1);
    float xv = x[((size_t)bb * Lc + il) * Dc + h * DKc + d];
    aE += xv * We[(h * LENc + l) * Rc + e];
    aR += xv * Wr[(h * LENc + l) * Rc + e];
  }
  we[t] = aE;
  wr[t] = aR;
}

// ---------------------------------------------------------------------------
// bf16 WMMA GEMM:  C(MxN) = A(MxK) @ Bt(NxK)^T + bias
//   128x128 block tile, BK=64, 256 threads = 8 waves.
//   Double-buffered LDS, async global->LDS staging overlapped with compute.
//   Each wave: 32x64 sub-tile -> 16 x v_wmma_f32_16x16x32_bf16 per K-step.
// ---------------------------------------------------------------------------
#define BM   128
#define BN   128
#define BKk  64
#define LDSS 72   // LDS row stride in bf16 elements (144 B, 16 B aligned)

template <bool OUT_BF16>
__global__ __launch_bounds__(256) void gemm_wmma_bf16(
    const unsigned short* __restrict__ A,   // M x K bf16 row-major
    const unsigned short* __restrict__ Bt,  // N x K bf16 row-major (B^T)
    const float* __restrict__ bias,         // N
    void* __restrict__ Cout,                // M x N (bf16 or f32)
    int M, int N, int K) {
  __shared__ __align__(16) unsigned short As[2][BM * LDSS];
  __shared__ __align__(16) unsigned short Bs[2][BN * LDSS];

  const int tid  = threadIdx.x;
  const int m0   = blockIdx.x * BM;
  const int n0   = blockIdx.y * BN;
  const int lane = tid & 31;
  const int wave = tid >> 5;
  const int wrow = (wave & 3) * 32;   // 4 row strips of 32
  const int wcol = (wave >> 2) * 64;  // 2 col strips of 64

  v8f acc[2][4];
#pragma unroll
  for (int i = 0; i < 2; ++i)
#pragma unroll
    for (int j = 0; j < 4; ++j)
      acc[i][j] = (v8f){0.f, 0.f, 0.f, 0.f, 0.f, 0.f, 0.f, 0.f};

  // staging: 2 threads per tile row, 32 elements (64 B) each -> 4 x b128
  const int sr = tid >> 1;
  const int sc = (tid & 1) * 32;
  // fragment addressing (ISA 16-bit A layout):
  //   lane<16: row fr, K = 0..7 & 16..23 ; lane>=16: K = 8..15 & 24..31
  const int half = lane >> 4;
  const int fr   = lane & 15;

  const unsigned short* ga = A  + (size_t)(m0 + sr) * K + sc;
  const unsigned short* gb = Bt + (size_t)(n0 + sr) * K + sc;

  auto stage = [&](int buf, int k0) {
#pragma unroll
    for (int c = 0; c < 4; ++c) {
      stage16(ga + k0 + c * 8, &As[buf][sr * LDSS + sc + c * 8]);
      stage16(gb + k0 + c * 8, &Bs[buf][sr * LDSS + sc + c * 8]);
    }
  };

  const int NT = K / BKk;
  stage(0, 0);
  stage_wait();
  __syncthreads();

  for (int kt = 0; kt < NT; ++kt) {
    const int cur = kt & 1;
    if (kt + 1 < NT) stage(cur ^ 1, (kt + 1) * BKk);  // overlap with compute

#pragma unroll
    for (int kk = 0; kk < BKk; kk += 32) {
      FragAB af[2], bfm[4];
#pragma unroll
      for (int mi = 0; mi < 2; ++mi) {
        const unsigned short* ap =
            &As[cur][(wrow + mi * 16 + fr) * LDSS + kk + half * 8];
        af[mi].q[0] = *(const uint4*)ap;
        af[mi].q[1] = *(const uint4*)(ap + 16);
      }
#pragma unroll
      for (int ni = 0; ni < 4; ++ni) {
        const unsigned short* bp =
            &Bs[cur][(wcol + ni * 16 + fr) * LDSS + kk + half * 8];
        bfm[ni].q[0] = *(const uint4*)bp;
        bfm[ni].q[1] = *(const uint4*)(bp + 16);
      }
#pragma unroll
      for (int mi = 0; mi < 2; ++mi)
#pragma unroll
        for (int ni = 0; ni < 4; ++ni)
          acc[mi][ni] = __builtin_amdgcn_wmma_f32_16x16x32_bf16(
              false, af[mi].v, false, bfm[ni].v, (short)0, acc[mi][ni],
              false, false);
    }

    if (kt + 1 < NT) stage_wait();  // next buffer landed
    __syncthreads();                // publish LDS + release cur for restaging
  }

  // Epilogue: C/D layout — VGPR v: lanes0-15 -> M=v, lanes16-31 -> M=v+8; N=lane&15
#pragma unroll
  for (int mi = 0; mi < 2; ++mi) {
#pragma unroll
    for (int ni = 0; ni < 4; ++ni) {
      int col = n0 + wcol + ni * 16 + fr;
      int rb  = m0 + wrow + mi * 16 + half * 8;
      float bv = bias[col];
#pragma unroll
      for (int v = 0; v < 8; ++v) {
        float val  = acc[mi][ni][v] + bv;
        size_t off = (size_t)(rb + v) * N + col;
        if (OUT_BF16) ((unsigned short*)Cout)[off] = f2bf(val);
        else          ((float*)Cout)[off] = val;
      }
    }
  }
}

// ---------------------------------------------------------------------------
// Score kernel: one wave per (row, head).
//   qn,kn = l2norm over dk=64 ; escore=qn@we(64x20), rscore=kn@wr ;
//   mid = [escore|rscore](40) @ Wc(40x64) + bc -> bf16
// ---------------------------------------------------------------------------
__global__ __launch_bounds__(256) void score_kernel(
    const unsigned short* __restrict__ qb,
    const unsigned short* __restrict__ kb,
    const float* __restrict__ we,
    const float* __restrict__ wr,
    const float* __restrict__ Wc,
    const float* __restrict__ bc,
    unsigned short* __restrict__ midb) {
  __shared__ float sQ[8][64];
  __shared__ float sK[8][64];
  __shared__ float sS[8][48];

  const int lane = threadIdx.x & 31;
  const int w    = threadIdx.x >> 5;
  const int task = blockIdx.x * 8 + w;  // row*16 + head
  const int row  = task >> 4;
  const int h    = task & 15;
  const int bb   = row >> 12;  // row / Lc

  const unsigned short* qs = qb + (size_t)row * Ndim + h * DKc;
  const unsigned short* ks = kb + (size_t)row * Ndim + h * DKc;
  float q0 = bf2f(qs[lane]), q1 = bf2f(qs[lane + 32]);
  float k0 = bf2f(ks[lane]), k1 = bf2f(ks[lane + 32]);
  float sq = q0 * q0 + q1 * q1;
  float sk = k0 * k0 + k1 * k1;
#pragma unroll
  for (int m = 16; m >= 1; m >>= 1) {
    sq += __shfl_xor(sq, m, 32);
    sk += __shfl_xor(sk, m, 32);
  }
  float qscale = 1.f / fmaxf(sqrtf(sq), 1e-12f);
  float kscale = 1.f / fmaxf(sqrtf(sk), 1e-12f);
  sQ[w][lane] = q0 * qscale; sQ[w][lane + 32] = q1 * qscale;
  sK[w][lane] = k0 * kscale; sK[w][lane + 32] = k1 * kscale;
  __syncthreads();

  if (lane < Rc) {
    const float* wep = we + (size_t)((bb * Hc + h) * DKc) * Rc + lane;
    const float* wrp = wr + (size_t)((bb * Hc + h) * DKc) * Rc + lane;
    float ae = 0.f, ar = 0.f;
#pragma unroll 8
    for (int d = 0; d < DKc; ++d) {
      ae += sQ[w][d] * wep[d * Rc];
      ar += sK[w][d] * wrp[d * Rc];
    }
    sS[w][lane]      = ae;
    sS[w][Rc + lane] = ar;
  }
  __syncthreads();

  float o0 = bc[lane], o1 = bc[lane + 32];
#pragma unroll 8
  for (int c = 0; c < 2 * Rc; ++c) {
    float s = sS[w][c];
    o0 += s * Wc[c * DKc + lane];
    o1 += s * Wc[c * DKc + lane + 32];
  }
  unsigned short* mp = midb + (size_t)row * Ndim + h * DKc;
  mp[lane]      = f2bf(o0);
  mp[lane + 32] = f2bf(o1);
}

// ---------------------------------------------------------------------------
// Launch
// ---------------------------------------------------------------------------
extern "C" void kernel_launch(void* const* d_in, const int* in_sizes, int n_in,
                              void* d_out, int out_size, void* d_ws,
                              size_t ws_size, hipStream_t stream) {
  (void)in_sizes; (void)n_in; (void)out_size; (void)ws_size;
  const float* x  = (const float*)d_in[0];
  const float* Wq = (const float*)d_in[1];
  const float* bq = (const float*)d_in[2];
  const float* Wk = (const float*)d_in[3];
  const float* bk = (const float*)d_in[4];
  const float* We = (const float*)d_in[5];
  const float* Wr = (const float*)d_in[6];
  const float* Wc = (const float*)d_in[7];
  const float* bc = (const float*)d_in[8];
  const float* Wo = (const float*)d_in[9];
  const float* bo = (const float*)d_in[10];

  // Workspace layout (bytes), all 256B aligned
  char* ws = (char*)d_ws;
  unsigned short* xb   = (unsigned short*)(ws);              // 32 MB
  unsigned short* qb2  = (unsigned short*)(ws + 33554432);   // 32 MB
  unsigned short* kb2  = (unsigned short*)(ws + 67108864);   // 32 MB
  unsigned short* midb = (unsigned short*)(ws + 100663296);  // 32 MB
  unsigned short* wqT  = (unsigned short*)(ws + 134217728);  // 2 MB
  unsigned short* wkT  = (unsigned short*)(ws + 136314880);  // 2 MB
  unsigned short* woT  = (unsigned short*)(ws + 138412032);  // 2 MB
  float* weP = (float*)(ws + 140509184);                     // 320 KB
  float* wrP = (float*)(ws + 140836864);                     // 320 KB

  // 1) conversions / transposes
  cvt_f32_bf16<<<(Mrows * Kdim) / 256, 256, 0, stream>>>(x, xb, Mrows * Kdim);
  cvt_transpose<<<(Kdim * Ndim) / 256, 256, 0, stream>>>(Wq, wqT, Kdim, Ndim);
  cvt_transpose<<<(Kdim * Ndim) / 256, 256, 0, stream>>>(Wk, wkT, Kdim, Ndim);
  cvt_transpose<<<(Kdim * Ndim) / 256, 256, 0, stream>>>(Wo, woT, Kdim, Ndim);

  // 2) landmark low-rank projections
  landmark_kernel<<<(Bc * Hc * DKc * Rc + 255) / 256, 256, 0, stream>>>(
      x, We, Wr, weP, wrP);

  // 3) big GEMMs via WMMA
  dim3 g(Mrows / BM, Ndim / BN);
  gemm_wmma_bf16<true><<<g, 256, 0, stream>>>(xb, wqT, bq, qb2, Mrows, Ndim, Kdim);
  gemm_wmma_bf16<true><<<g, 256, 0, stream>>>(xb, wkT, bk, kb2, Mrows, Ndim, Kdim);

  // 4) normalize + score + Wc projection (VALU; ~1.3% of FLOPs)
  score_kernel<<<(Bc * Lc * Hc) / 8, 256, 0, stream>>>(qb2, kb2, weP, wrP, Wc,
                                                       bc, midb);

  // 5) output projection -> f32 d_out
  gemm_wmma_bf16<false><<<g, 256, 0, stream>>>(midb, woT, bo, d_out, Mrows,
                                               Ndim, Kdim);
}
